// BertSelfAttention_11450382812023
// MI455X (gfx1250) — matrix operand order repaired and checked by reading, not compiled, not run on hardware
//
#include <hip/hip_runtime.h>
#include <hip/hip_bf16.h>
#include <stdint.h>

// Problem constants: B=1, S=384, H=768, NH=12, D=64
#define S_LEN 384
#define H_DIM 768
#define NHEAD 12
#define DHEAD 64
#define NHP   16   // heads padded to 16 for WMMA N dimension

typedef __bf16 bf16_t;
typedef __attribute__((ext_vector_type(16))) __bf16 v16bf;
typedef __attribute__((ext_vector_type(8)))  __bf16 bf16x8;
typedef __attribute__((ext_vector_type(8)))  float   v8f;
typedef __attribute__((ext_vector_type(4)))  float   f32x4;

// Async global->LDS path (gfx1250): direct cache->LDS copy on ASYNCcnt,
// no VGPR landing zone.  Guarded so the file compiles on toolchains that
// lack the builtin (falls back to a plain copy loop).
#if defined(__has_builtin)
#  if __has_builtin(__builtin_amdgcn_global_load_async_to_lds_b128)
#    define HAVE_ASYNC_LDS 1
#  endif
#endif

#ifdef HAVE_ASYNC_LDS
// Builtin signature (learned from clang diagnostics): 4 args,
//   arg0: addrspace(1) int4*  (global source)
//   arg1: addrspace(3) int4*  (LDS destination)
//   arg2: imm offset, arg3: cpol
typedef int v4i __attribute__((ext_vector_type(4)));
typedef __attribute__((address_space(1))) v4i as1_v4i;
typedef __attribute__((address_space(3))) v4i as3_v4i;
#endif

// ---------------------------------------------------------------------------
// WMMA helper: D = A(16x32 bf16) * B(32x16 bf16) + C(16x16 f32)
// ---------------------------------------------------------------------------
static __device__ __forceinline__ v8f wmma_bf16(v16bf a, v16bf b, v8f c) {
  return __builtin_amdgcn_wmma_f32_16x16x32_bf16(
      /*neg_a=*/false, a, /*neg_b=*/false, b,
      /*c_mod=*/(short)0, c, /*reuse_a=*/false, /*reuse_b=*/false);
}

// A-operand loader, 16x32 bf16 tile, row-major source.
// ISA layout: lane m (=l&15) holds row m; K chunks: (l>>4)*8 + {0..7, 16..23}.
// p must already be:  row_base(lane) + (lane>>4)*8
static __device__ __forceinline__ v16bf ldA(const bf16_t* p) {
  bf16x8 c0 = *(const bf16x8*)(p);
  bf16x8 c1 = *(const bf16x8*)(p + 16);
  v16bf a;
#pragma unroll
  for (int j = 0; j < 8; ++j) { a[j] = c0[j]; a[8 + j] = c1[j]; }
  return a;
}

// B-operand loader, 32x16 bf16 tile, column-contiguous source.
// Lane n (=l&15) holds column n; K chunks: (l>>4)*8 + {0..7, 16..23}.
// p must already be:  col_base(lane&15) + (lane>>4)*8
static __device__ __forceinline__ v16bf ldB(const bf16_t* p) {
  bf16x8 c0 = *(const bf16x8*)(p);
  bf16x8 c1 = *(const bf16x8*)(p + 16);
  v16bf b;
#pragma unroll
  for (int j = 0; j < 8; ++j) { b[j] = c0[j]; b[8 + j] = c1[j]; }
  return b;
}

// ---------------------------------------------------------------------------
// K0: f32 -> bf16 copy, and Wr transpose (Wrt[i][o] = Wr[o][i])
// ---------------------------------------------------------------------------
__global__ void cvt_kernel(const float* __restrict__ src, bf16_t* __restrict__ dst, int n) {
  int i = blockIdx.x * 256 + threadIdx.x;
  if (i < n) dst[i] = (bf16_t)src[i];
}

__global__ void transpose_kernel(const float* __restrict__ src, bf16_t* __restrict__ dst) {
  int i = blockIdx.x * 256 + threadIdx.x;
  if (i < H_DIM * H_DIM) {
    int o = i / H_DIM, in = i % H_DIM;
    dst[in * H_DIM + o] = (bf16_t)src[o * H_DIM + in];
  }
}

// ---------------------------------------------------------------------------
// K1: QKV projections.  out[s,o] = X[s,:] . W[o,:] + b[o]
//   z=0 -> Qu = q + u  and  Qv = q + v   (bf16, [h][s][d])
//   z=1 -> K                              (bf16, [h][s][d])
//   z=2 -> V transposed                   (bf16, [h][d][s])
// grid = (24 s-tiles, 48 o-tiles, 3), block = 32 (one wave)
// ---------------------------------------------------------------------------
__global__ void __launch_bounds__(32)
proj_qkv_kernel(const bf16_t* __restrict__ Xb,
                const bf16_t* __restrict__ Wqb, const bf16_t* __restrict__ Wkb,
                const bf16_t* __restrict__ Wvb,
                const float* __restrict__ bq, const float* __restrict__ bk,
                const float* __restrict__ bv,
                const float* __restrict__ u, const float* __restrict__ v,
                bf16_t* __restrict__ Qu, bf16_t* __restrict__ Qv,
                bf16_t* __restrict__ Kb, bf16_t* __restrict__ Vt) {
  const int mt = blockIdx.x, nt = blockIdx.y, z = blockIdx.z;
  const int l = threadIdx.x, lo = l & 15, hi = l >> 4;
  const int s0 = mt * 16, o0 = nt * 16;
  const bf16_t* W   = (z == 0) ? Wqb : (z == 1) ? Wkb : Wvb;
  const float*  bia = (z == 0) ? bq  : (z == 1) ? bk  : bv;

  const bf16_t* arow = Xb + (size_t)(s0 + lo) * H_DIM + hi * 8;
  const bf16_t* bcol = W  + (size_t)(o0 + lo) * H_DIM + hi * 8;
  v8f acc = {};
#pragma unroll 4
  for (int k0 = 0; k0 < H_DIM; k0 += 32)
    acc = wmma_bf16(ldA(arow + k0), ldB(bcol + k0), acc);

  const int o = o0 + lo, h = o >> 6, d = o & 63;
#pragma unroll
  for (int r = 0; r < 8; ++r) {
    const int s = s0 + r + 8 * hi;
    const float val = acc[r] + bia[o];
    if (z == 0) {
      Qu[((size_t)h * S_LEN + s) * DHEAD + d] = (bf16_t)(val + u[o]);
      Qv[((size_t)h * S_LEN + s) * DHEAD + d] = (bf16_t)(val + v[o]);
    } else if (z == 1) {
      Kb[((size_t)h * S_LEN + s) * DHEAD + d] = (bf16_t)val;
    } else {
      Vt[((size_t)h * DHEAD + d) * S_LEN + s] = (bf16_t)val;
    }
  }
}

// ---------------------------------------------------------------------------
// K2: P[q,h,i] = sum_d Qv[h,q,d] * Wr[h*64+d, i]   (via Wrt[i][o])
// Output layout P[q][16][768] bf16 (heads 12..15 left unwritten; their B
// columns feed garbage into WMMA lanes whose results are discarded).
// grid = (12 heads, 24 q-tiles, 48 i-tiles), block = 32
// ---------------------------------------------------------------------------
__global__ void __launch_bounds__(32)
p_kernel(const bf16_t* __restrict__ Qv, const bf16_t* __restrict__ Wrt,
         bf16_t* __restrict__ P) {
  const int h = blockIdx.x, qt = blockIdx.y, it = blockIdx.z;
  const int l = threadIdx.x, lo = l & 15, hi = l >> 4;
  const int q0 = qt * 16, i0 = it * 16;

  const bf16_t* arow = Qv  + ((size_t)h * S_LEN + q0 + lo) * DHEAD + hi * 8;
  const bf16_t* bcol = Wrt + (size_t)(i0 + lo) * H_DIM + h * DHEAD + hi * 8;
  v8f acc = {};
  acc = wmma_bf16(ldA(arow),      ldB(bcol),      acc);
  acc = wmma_bf16(ldA(arow + 32), ldB(bcol + 32), acc);

#pragma unroll
  for (int r = 0; r < 8; ++r) {
    const int q = q0 + r + 8 * hi;
    P[((size_t)q * NHP + h) * H_DIM + (i0 + lo)] = (bf16_t)acc[r];
  }
}

// ---------------------------------------------------------------------------
// K3: a_c[q,k,h] = Qu[h,q,:] . K[h,k,:]   stored [q][k][16] f32
// grid = (12 heads, 24 q-tiles, 24 k-tiles), block = 32
// ---------------------------------------------------------------------------
__global__ void __launch_bounds__(32)
ac_kernel(const bf16_t* __restrict__ Qu, const bf16_t* __restrict__ Kb,
          float* __restrict__ AC) {
  const int h = blockIdx.x, qt = blockIdx.y, kt = blockIdx.z;
  const int l = threadIdx.x, lo = l & 15, hi = l >> 4;
  const int q0 = qt * 16, k0 = kt * 16;

  const bf16_t* arow = Qu + ((size_t)h * S_LEN + q0 + lo) * DHEAD + hi * 8;
  const bf16_t* bcol = Kb + ((size_t)h * S_LEN + k0 + lo) * DHEAD + hi * 8;
  v8f acc = {};
  acc = wmma_bf16(ldA(arow),      ldB(bcol),      acc);
  acc = wmma_bf16(ldA(arow + 32), ldB(bcol + 32), acc);

#pragma unroll
  for (int r = 0; r < 8; ++r) {
    const int q = q0 + r + 8 * hi;
    AC[((size_t)q * S_LEN + (k0 + lo)) * NHP + h] = acc[r];
  }
}

// ---------------------------------------------------------------------------
// K4: cbr[h,q] = Qv[h,q,:] . br[h*64 : h*64+64]   (exactness for the br bias)
// ---------------------------------------------------------------------------
__global__ void cbr_kernel(const bf16_t* __restrict__ Qv, const float* __restrict__ br,
                           float* __restrict__ cbr) {
  int i = blockIdx.x * 256 + threadIdx.x;
  if (i < NHEAD * S_LEN) {
    int h = i / S_LEN, q = i % S_LEN;
    float s = 0.f;
#pragma unroll 8
    for (int d = 0; d < DHEAD; ++d)
      s += (float)Qv[((size_t)h * S_LEN + q) * DHEAD + d] * br[h * DHEAD + d];
    cbr[i] = s;
  }
}

// ---------------------------------------------------------------------------
// K5: fused relative-attention pass.  One workgroup per query q:
//   b_d[k,h] = rel_emb[q] (384x768, streamed once) x P[q] (768x16, in LDS)
//   scores   = (a_c + b_d + cbr)/8 + mask ; stable softmax over k
//   probs -> PR[h][q][k] bf16
// grid = 384, block = 256 (8 waves; each wave owns 3 of 24 k-tiles)
// ---------------------------------------------------------------------------
__global__ void __launch_bounds__(256)
relattn_kernel(const float* __restrict__ rel, const float* __restrict__ AC,
               const bf16_t* __restrict__ P, const float* __restrict__ cbr,
               const float* __restrict__ mask, bf16_t* __restrict__ PR) {
  __shared__ bf16_t Pl[NHP * H_DIM];      // 24 KB  (P[q], B-operand)
  __shared__ float  Sc[S_LEN * NHP];      // 24 KB  scores [k][h]
  __shared__ float  pmax[16][16], psum[16][16], gmax[16], gsum[16];

  const int q = blockIdx.x;
  const int t = threadIdx.x;

  { // stage P[q] (24 KB) into LDS
#ifdef HAVE_ASYNC_LDS
    // global_load_async_to_lds_b128: per-lane 16B cache->LDS, ASYNCcnt-tracked
    const bf16_t* gsrc = P + (size_t)q * NHP * H_DIM;
#pragma unroll
    for (int i = t; i < NHP * H_DIM / 8; i += 256) {   // 1536 x b128
      __builtin_amdgcn_global_load_async_to_lds_b128(
          (as1_v4i*)(uintptr_t)(gsrc + i * 8),
          (as3_v4i*)(Pl + i * 8), 0, 0);
    }
#  if __has_builtin(__builtin_amdgcn_s_wait_asynccnt)
    __builtin_amdgcn_s_wait_asynccnt(0);
#  else
    asm volatile("s_wait_asynccnt 0" ::: "memory");
#  endif
#else
    const uint32_t* src = (const uint32_t*)(P + (size_t)q * NHP * H_DIM);
    uint32_t* dst = (uint32_t*)Pl;
#pragma unroll
    for (int i = t; i < NHP * H_DIM / 2; i += 256) dst[i] = src[i];
#endif
  }
  __syncthreads();

  const int w = t >> 5, l = t & 31, lo = l & 15, hi = l >> 4;

  for (int mi = w; mi < 24; mi += 8) {
    const int k0 = mi * 16;
    const float*  arow = rel + ((size_t)q * S_LEN + k0 + lo) * H_DIM + hi * 8;
    const bf16_t* bcol = Pl + lo * H_DIM + hi * 8;
    v8f acc = {};
    for (int e0 = 0; e0 < H_DIM; e0 += 32) {
      __builtin_prefetch(arow + e0 + 256, 0, 1);   // global_prefetch_b8
      f32x4 c0 = *(const f32x4*)(arow + e0);
      f32x4 c1 = *(const f32x4*)(arow + e0 + 4);
      f32x4 c2 = *(const f32x4*)(arow + e0 + 16);
      f32x4 c3 = *(const f32x4*)(arow + e0 + 20);
      v16bf a;
#pragma unroll
      for (int j = 0; j < 4; ++j) {
        a[j]      = (bf16_t)c0[j];  a[4 + j]  = (bf16_t)c1[j];
        a[8 + j]  = (bf16_t)c2[j];  a[12 + j] = (bf16_t)c3[j];
      }
      acc = wmma_bf16(a, ldB(bcol + e0), acc);
    }
    const float cb = cbr[(lo < NHEAD ? lo : 0) * S_LEN + q];
#pragma unroll
    for (int r = 0; r < 8; ++r) {
      const int k = k0 + r + 8 * hi;
      const float sc = (acc[r] + AC[((size_t)q * S_LEN + k) * NHP + lo] + cb) * 0.125f
                       + mask[k];
      Sc[k * NHP + lo] = sc;
    }
  }
  __syncthreads();

  { // partial max over 16 k-segments of 24
    const int h = t & 15, seg = t >> 4;
    float m = -1e30f;
    for (int j = 0; j < 24; ++j) m = fmaxf(m, Sc[(seg * 24 + j) * NHP + h]);
    pmax[seg][h] = m;
  }
  __syncthreads();
  if (t < 16) {
    float m = -1e30f;
    for (int sgi = 0; sgi < 16; ++sgi) m = fmaxf(m, pmax[sgi][t]);
    gmax[t] = m;
  }
  __syncthreads();
  {
    const int h = t & 15, seg = t >> 4;
    const float m = gmax[h];
    float s = 0.f;
    for (int j = 0; j < 24; ++j) s += __expf(Sc[(seg * 24 + j) * NHP + h] - m);
    psum[seg][h] = s;
  }
  __syncthreads();
  if (t < 16) {
    float s = 0.f;
    for (int sgi = 0; sgi < 16; ++sgi) s += psum[sgi][t];
    gsum[t] = s;
  }
  __syncthreads();

  for (int idx = t; idx < NHEAD * S_LEN; idx += 256) {
    const int h = idx / S_LEN, k = idx % S_LEN;
    const float p = __expf(Sc[k * NHP + h] - gmax[h]) / gsum[h];
    PR[((size_t)h * S_LEN + q) * S_LEN + k] = (bf16_t)p;
  }
}

// ---------------------------------------------------------------------------
// K6: ctx[q, h*64+d] = sum_k probs[h,q,k] * V[h,k,d]   (Vt is [h][d][k])
// grid = (12 heads, 24 q-tiles, 4 d-tiles), block = 32
// ---------------------------------------------------------------------------
__global__ void __launch_bounds__(32)
ctx_kernel(const bf16_t* __restrict__ PR, const bf16_t* __restrict__ Vt,
           float* __restrict__ out) {
  const int h = blockIdx.x, qt = blockIdx.y, dt = blockIdx.z;
  const int l = threadIdx.x, lo = l & 15, hi = l >> 4;
  const int q0 = qt * 16, d0 = dt * 16;

  const bf16_t* arow = PR + ((size_t)h * S_LEN + q0 + lo) * S_LEN + hi * 8;
  const bf16_t* bcol = Vt + ((size_t)h * DHEAD + d0 + lo) * S_LEN + hi * 8;
  v8f acc = {};
#pragma unroll 4
  for (int k0 = 0; k0 < S_LEN; k0 += 32)
    acc = wmma_bf16(ldA(arow + k0), ldB(bcol + k0), acc);

#pragma unroll
  for (int r = 0; r < 8; ++r) {
    const int qq = q0 + r + 8 * hi;
    out[(size_t)qq * H_DIM + h * DHEAD + d0 + lo] = acc[r];
  }
}

// ---------------------------------------------------------------------------
extern "C" void kernel_launch(void* const* d_in, const int* in_sizes, int n_in,
                              void* d_out, int out_size, void* d_ws, size_t ws_size,
                              hipStream_t stream) {
  const float* hs   = (const float*)d_in[0];
  const float* rel  = (const float*)d_in[1];
  const float* mask = (const float*)d_in[2];
  const float* Wq   = (const float*)d_in[3];
  const float* bq   = (const float*)d_in[4];
  const float* Wk   = (const float*)d_in[5];
  const float* bk   = (const float*)d_in[6];
  const float* Wv   = (const float*)d_in[7];
  const float* bv   = (const float*)d_in[8];
  const float* Wr   = (const float*)d_in[9];
  const float* br   = (const float*)d_in[10];
  const float* u    = (const float*)d_in[11];
  const float* v    = (const float*)d_in[12];
  float* out = (float*)d_out;

  // workspace carve-up (256B aligned blocks)
  uint8_t* ws = (uint8_t*)d_ws;
  size_t o = 0;
  auto take = [&](size_t bytes) { uint8_t* p = ws + o; o += (bytes + 255) & ~(size_t)255; return p; };
  bf16_t* Xb  = (bf16_t*)take((size_t)S_LEN * H_DIM * 2);
  bf16_t* Wqb = (bf16_t*)take((size_t)H_DIM * H_DIM * 2);
  bf16_t* Wkb = (bf16_t*)take((size_t)H_DIM * H_DIM * 2);
  bf16_t* Wvb = (bf16_t*)take((size_t)H_DIM * H_DIM * 2);
  bf16_t* Wrt = (bf16_t*)take((size_t)H_DIM * H_DIM * 2);
  bf16_t* Qu  = (bf16_t*)take((size_t)S_LEN * H_DIM * 2);
  bf16_t* Qv  = (bf16_t*)take((size_t)S_LEN * H_DIM * 2);
  bf16_t* Kb  = (bf16_t*)take((size_t)S_LEN * H_DIM * 2);
  bf16_t* Vt  = (bf16_t*)take((size_t)S_LEN * H_DIM * 2);
  bf16_t* P   = (bf16_t*)take((size_t)S_LEN * NHP * H_DIM * 2);
  float*  AC  = (float*) take((size_t)S_LEN * S_LEN * NHP * 4);
  bf16_t* PR  = (bf16_t*)take((size_t)NHEAD * S_LEN * S_LEN * 2);
  float*  CBR = (float*) take((size_t)NHEAD * S_LEN * 4);

  // stage 0: precision conversion
  cvt_kernel<<<(S_LEN * H_DIM + 255) / 256, 256, 0, stream>>>(hs, Xb, S_LEN * H_DIM);
  cvt_kernel<<<(H_DIM * H_DIM + 255) / 256, 256, 0, stream>>>(Wq, Wqb, H_DIM * H_DIM);
  cvt_kernel<<<(H_DIM * H_DIM + 255) / 256, 256, 0, stream>>>(Wk, Wkb, H_DIM * H_DIM);
  cvt_kernel<<<(H_DIM * H_DIM + 255) / 256, 256, 0, stream>>>(Wv, Wvb, H_DIM * H_DIM);
  transpose_kernel<<<(H_DIM * H_DIM + 255) / 256, 256, 0, stream>>>(Wr, Wrt);

  // stage 1: QKV projections (WMMA)
  proj_qkv_kernel<<<dim3(S_LEN / 16, H_DIM / 16, 3), 32, 0, stream>>>(
      Xb, Wqb, Wkb, Wvb, bq, bk, bv, u, v, Qu, Qv, Kb, Vt);

  // stage 2: P, a_c, cbr
  p_kernel<<<dim3(NHEAD, S_LEN / 16, H_DIM / 16), 32, 0, stream>>>(Qv, Wrt, P);
  ac_kernel<<<dim3(NHEAD, S_LEN / 16, S_LEN / 16), 32, 0, stream>>>(Qu, Kb, AC);
  cbr_kernel<<<(NHEAD * S_LEN + 255) / 256, 256, 0, stream>>>(Qv, br, CBR);

  // stage 3: fused single-pass rel_embedding stream + softmax
  relattn_kernel<<<S_LEN, 256, 0, stream>>>(rel, AC, P, CBR, mask, PR);

  // stage 4: context
  ctx_kernel<<<dim3(NHEAD, S_LEN / 16, DHEAD / 16), 32, 0, stream>>>(PR, Vt, out);
}
